// ScaledDotProductAttention_19396072309378
// MI455X (gfx1250) — matrix operand order repaired and checked by reading, not compiled, not run on hardware
//
#include <hip/hip_runtime.h>

// Flash-attention forward, causal, bf16 WMMA / fp32 softmax, gfx1250 wave32.
// Row-max via immediate ds_swizzle (no address VALU), row-sum via WMMA x ones,
// exp2-domain softmax (log2e folded into Q scale), bf16 LDS tiles, double-buffered.

typedef __attribute__((ext_vector_type(16))) __bf16 v16bf;
typedef __attribute__((ext_vector_type(8)))  __bf16 v8bf;
typedef __attribute__((ext_vector_type(8)))  float  v8f;

#define SQ      2048
#define DK      128
#define BH      64      // B*H
#define BLOCK_M 128     // query rows per workgroup (8 waves x 16)
#define BLOCK_N 32      // key rows per iteration
#define NWAVE   8
#define KPAD    (DK + 8)       // bf16 per K-tile row (272 B)
#define VPAD    (BLOCK_N + 8)  // bf16 per Vt row (80 B)
#define PPAD    (BLOCK_N + 8)  // bf16 per P row (80 B, 16B-aligned runs)

#if __has_builtin(__builtin_amdgcn_exp2f)
#define EXP2(x) __builtin_amdgcn_exp2f(x)
#else
#define EXP2(x) exp2f(x)
#endif

// xor-max over 16-lane groups, zero-VALU lane exchange via ds_swizzle imm.
__device__ __forceinline__ float xmax16(float t) {
  t = fmaxf(t, __int_as_float(__builtin_amdgcn_ds_swizzle(__float_as_int(t), 0x041f)));
  t = fmaxf(t, __int_as_float(__builtin_amdgcn_ds_swizzle(__float_as_int(t), 0x081f)));
  t = fmaxf(t, __int_as_float(__builtin_amdgcn_ds_swizzle(__float_as_int(t), 0x101f)));
  t = fmaxf(t, __int_as_float(__builtin_amdgcn_ds_swizzle(__float_as_int(t), 0x201f)));
  return t;
}

// A-operand fragment (16x32 bf16) from f32 global, scale folded in (Q only).
__device__ __forceinline__ v16bf fragA_f32s(const float* __restrict__ row, int half, float sc) {
  const float* a = row + half * 8;
  const float* b = row + 16 + half * 8;
  v16bf r;
#pragma unroll
  for (int j = 0; j < 8; ++j) {
    r[j]     = (__bf16)(a[j] * sc);
    r[j + 8] = (__bf16)(b[j] * sc);
  }
  return r;
}

// A-operand fragment from bf16 LDS (P tiles): two aligned 16-B runs.
__device__ __forceinline__ v16bf fragA_bf16(const __bf16* __restrict__ row, int half) {
  const __bf16* a = row + half * 8;
  const __bf16* b = row + 16 + half * 8;
  v16bf r;
#pragma unroll
  for (int j = 0; j < 8; ++j) {
    r[j]     = a[j];
    r[j + 8] = b[j];
  }
  return r;
}

// B-operand fragment (32x16 bf16) from bf16 LDS: two aligned 16-B runs.
__device__ __forceinline__ v16bf fragB_bf16(const __bf16* __restrict__ row, int half) {
  const __bf16* a = row + half * 8;
  const __bf16* b = row + 16 + half * 8;
  v16bf r;
#pragma unroll
  for (int j = 0; j < 8; ++j) {
    r[j]     = a[j];
    r[j + 8] = b[j];
  }
  return r;
}

__device__ __forceinline__ v8bf pack8(float4 a, float4 b) {
  v8bf r;
  r[0] = (__bf16)a.x; r[1] = (__bf16)a.y; r[2] = (__bf16)a.z; r[3] = (__bf16)a.w;
  r[4] = (__bf16)b.x; r[5] = (__bf16)b.y; r[6] = (__bf16)b.z; r[7] = (__bf16)b.w;
  return r;
}

// Online-softmax (exp2 domain): mask, row-max, alpha, P->LDS. Sum done via WMMA.
template <bool MASKED>
__device__ __forceinline__ void sm_update(const v8f& s0, const v8f& s1,
                                          float* __restrict__ mrow, float* __restrict__ alpha,
                                          __bf16* __restrict__ Prow,
                                          int qbase, int kn0, int half, int ln) {
  const int key0 = kn0 + ln;
  const int key1 = kn0 + 16 + ln;
#pragma unroll
  for (int r = 0; r < 8; ++r) {
    const int qrow = qbase + half * 8 + r;
    float x0 = s0[r];
    float x1 = s1[r];
    if (MASKED) {
      if (key0 > qrow) x0 = -1.0e30f;
      if (key1 > qrow) x1 = -1.0e30f;
    }
    const float t  = xmax16(fmaxf(x0, x1));
    const float mn = fmaxf(mrow[r], t);
    alpha[r] = EXP2(mrow[r] - mn);
    mrow[r]  = mn;
    const float p0 = EXP2(x0 - mn);
    const float p1 = EXP2(x1 - mn);
    __bf16* p = Prow + (size_t)(half * 8 + r) * PPAD;
    p[ln]      = (__bf16)p0;
    p[ln + 16] = (__bf16)p1;
  }
}

__global__ __launch_bounds__(256, 1)
void fa_fwd_causal_kernel(const float* __restrict__ Q, const float* __restrict__ K,
                          const float* __restrict__ V, float* __restrict__ O) {
  __shared__ __bf16 Kbf[2][BLOCK_N][KPAD];   // K tiles, row-major bf16
  __shared__ __bf16 Vtb[2][DK][VPAD];        // V tiles, transposed (d-major) bf16
  __shared__ __bf16 Plds[NWAVE][16][PPAD];   // per-wave P staging

  const int bh   = blockIdx.x / (SQ / BLOCK_M);
  const int qb   = blockIdx.x % (SQ / BLOCK_M);
  const int tid  = threadIdx.x;
  const int w    = tid >> 5;
  const int lane = tid & 31;
  const int half = lane >> 4;
  const int ln   = lane & 15;

  const size_t base = (size_t)bh * SQ * DK;
  const float* Qb = Q + base;
  const float* Kb = K + base;
  const float* Vb = V + base;
  float*       Ob = O + base;

  // Staging assignment: thread -> (key row, 16-float segment of the 128-d row).
  const int srow = tid >> 3;  // 0..31
  const int sseg = tid & 7;   // 0..7

  const int   qrow0  = qb * BLOCK_M + w * 16;
  const float qscale = 0.08838834764831845f * 1.4426950408889634f; // log2(e)/sqrt(128)

  // Q tile (16x128) as 4 bf16 A-fragments, scale pre-folded (one-time cost).
  v16bf qf[4];
#pragma unroll
  for (int kc = 0; kc < 4; ++kc)
    qf[kc] = fragA_f32s(Qb + (size_t)(qrow0 + ln) * DK + kc * 32, half, qscale);

  // Constant all-ones B fragment: WMMA row-sum of P.
  v16bf onesb;
#pragma unroll
  for (int j = 0; j < 16; ++j) onesb[j] = (__bf16)1.0f;

  v8f zero = {};
  v8f acc[8];
#pragma unroll
  for (int i = 0; i < 8; ++i) acc[i] = zero;
  float mrow[8], lrow[8];
#pragma unroll
  for (int r = 0; r < 8; ++r) { mrow[r] = -3.0e38f; lrow[r] = 0.0f; }

  const int nkt = 4 * qb + 4;  // key tiles covering the block (causal)

  float4 kreg[4], vreg[4];

  // --- prologue: fetch + commit tile 0 into buffer 0 ---
  {
    const float* kp = Kb + (size_t)srow * DK + sseg * 16;
    const float* vp = Vb + (size_t)srow * DK + sseg * 16;
#pragma unroll
    for (int i = 0; i < 4; ++i) { kreg[i] = ((const float4*)kp)[i]; vreg[i] = ((const float4*)vp)[i]; }
    *(v8bf*)&Kbf[0][srow][sseg * 16]     = pack8(kreg[0], kreg[1]);
    *(v8bf*)&Kbf[0][srow][sseg * 16 + 8] = pack8(kreg[2], kreg[3]);
#pragma unroll
    for (int i = 0; i < 4; ++i) {
      const float4 f = vreg[i];
      Vtb[0][sseg * 16 + i * 4 + 0][srow] = (__bf16)f.x;
      Vtb[0][sseg * 16 + i * 4 + 1][srow] = (__bf16)f.y;
      Vtb[0][sseg * 16 + i * 4 + 2][srow] = (__bf16)f.z;
      Vtb[0][sseg * 16 + i * 4 + 3][srow] = (__bf16)f.w;
    }
  }
  __syncthreads();

  for (int kt = 0; kt < nkt; ++kt) {
    const int  kn0       = kt * BLOCK_N;
    const int  cur       = kt & 1;
    const bool have_next = (kt + 1) < nkt;

    // Issue next tile's global loads up front (latency hidden behind WMMAs).
    if (have_next) {
      const float* kp = Kb + (size_t)(kn0 + BLOCK_N + srow) * DK + sseg * 16;
      const float* vp = Vb + (size_t)(kn0 + BLOCK_N + srow) * DK + sseg * 16;
#pragma unroll
      for (int i = 0; i < 4; ++i) { kreg[i] = ((const float4*)kp)[i]; vreg[i] = ((const float4*)vp)[i]; }
    }
    if (kt + 2 < nkt) {
      __builtin_prefetch(Kb + (size_t)(kn0 + 2 * BLOCK_N + srow) * DK + sseg * 16, 0, 1);
      __builtin_prefetch(Vb + (size_t)(kn0 + 2 * BLOCK_N + srow) * DK + sseg * 16, 0, 1);
    }

    // Skip tiles entirely above the diagonal for this wave (true causal work).
    const bool active = kn0 <= qrow0 + 15;
    if (active) {
      // S = Q K^T : two 16x16 tiles (32 keys), K-dim 128 in 4 chunks.
      v8f s0 = zero, s1 = zero;
#pragma unroll
      for (int kc = 0; kc < 4; ++kc) {
        v16bf kb0 = fragB_bf16(&Kbf[cur][ln][kc * 32], half);
        v16bf kb1 = fragB_bf16(&Kbf[cur][16 + ln][kc * 32], half);
        s0 = __builtin_amdgcn_wmma_f32_16x16x32_bf16(false, qf[kc], false, kb0,
                                                     (short)0, s0, false, false);
        s1 = __builtin_amdgcn_wmma_f32_16x16x32_bf16(false, qf[kc], false, kb1,
                                                     (short)0, s1, false, false);
      }

      float alpha[8];
      const bool needmask = (kn0 + 31) > qrow0;  // wave's diagonal tile
      if (needmask) {
        sm_update<true>(s0, s1, mrow, alpha, &Plds[w][0][0], qrow0, kn0, half, ln);
      } else {
        sm_update<false>(s0, s1, mrow, alpha, &Plds[w][0][0], qrow0, kn0, half, ln);
      }

      // Reload P as A-fragment (same-wave DS round-trip; in-order).
      v16bf pf = fragA_bf16(&Plds[w][ln][0], half);

      // Row sums of P via WMMA against ones: every column slot holds its row's sum.
      v8f sumt = __builtin_amdgcn_wmma_f32_16x16x32_bf16(false, pf, false, onesb,
                                                         (short)0, zero, false, false);
#pragma unroll
      for (int r = 0; r < 8; ++r) lrow[r] = lrow[r] * alpha[r] + sumt[r];
#pragma unroll
      for (int nd = 0; nd < 8; ++nd)
#pragma unroll
        for (int r = 0; r < 8; ++r) acc[nd][r] *= alpha[r];

      // O += P V : V-fragments from transposed LDS tile.
#pragma unroll
      for (int nd = 0; nd < 8; ++nd) {
        v16bf vf = fragB_bf16(&Vtb[cur][nd * 16 + ln][0], half);
        acc[nd] = __builtin_amdgcn_wmma_f32_16x16x32_bf16(false, pf, false, vf,
                                                          (short)0, acc[nd], false, false);
      }
    }

    // Commit next tile to the other buffer, then one barrier per iteration.
    if (have_next) {
      const int nxt = cur ^ 1;
      *(v8bf*)&Kbf[nxt][srow][sseg * 16]     = pack8(kreg[0], kreg[1]);
      *(v8bf*)&Kbf[nxt][srow][sseg * 16 + 8] = pack8(kreg[2], kreg[3]);
#pragma unroll
      for (int i = 0; i < 4; ++i) {
        const float4 f = vreg[i];
        Vtb[nxt][sseg * 16 + i * 4 + 0][srow] = (__bf16)f.x;
        Vtb[nxt][sseg * 16 + i * 4 + 1][srow] = (__bf16)f.y;
        Vtb[nxt][sseg * 16 + i * 4 + 2][srow] = (__bf16)f.z;
        Vtb[nxt][sseg * 16 + i * 4 + 3][srow] = (__bf16)f.w;
      }
    }
    __syncthreads();
  }

  // Epilogue: O = acc / l, coalesced f32 stores.
#pragma unroll
  for (int r = 0; r < 8; ++r) {
    const float inv = 1.0f / lrow[r];
    const int qrow  = qrow0 + half * 8 + r;
#pragma unroll
    for (int nd = 0; nd < 8; ++nd)
      Ob[(size_t)qrow * DK + nd * 16 + ln] = acc[nd][r] * inv;
  }
}

extern "C" void kernel_launch(void* const* d_in, const int* in_sizes, int n_in,
                              void* d_out, int out_size, void* d_ws, size_t ws_size,
                              hipStream_t stream) {
  (void)in_sizes; (void)n_in; (void)d_ws; (void)ws_size; (void)out_size;
  const float* Q = (const float*)d_in[0];
  const float* K = (const float*)d_in[1];
  const float* V = (const float*)d_in[2];
  float*       O = (float*)d_out;
  dim3 grid(BH * (SQ / BLOCK_M)); // 1024 workgroups
  dim3 block(256);                // 8 wave32
  hipLaunchKernelGGL(fa_fwd_causal_kernel, grid, block, 0, stream, Q, K, V, O);
}
